// MoEExperts_27109833572673
// MI455X (gfx1250) — matrix operand order repaired and checked by reading, not compiled, not run on hardware
//
#include <hip/hip_runtime.h>
#include <stdint.h>

// ---------------- problem constants (match reference) ----------------
#define E_   32
#define H_   512
#define I_   1024
#define T_   4096
#define K_   2
#define CAP_ 512   // 2*(T*K)/E

// ---------------- CDNA5 WMMA types ----------------
typedef __attribute__((ext_vector_type(16))) __bf16 bf16x16;
typedef __attribute__((ext_vector_type(8)))  float  f32x8;

union FragB { bf16x16 v; uint4 q[2]; };

__device__ __forceinline__ unsigned short f2bf_rne(float f) {
  unsigned int u = __float_as_uint(f);
  u += 0x7FFFu + ((u >> 16) & 1u);           // round-to-nearest-even
  return (unsigned short)(u >> 16);
}
__device__ __forceinline__ float bf2f(unsigned short h) {
  return __uint_as_float(((unsigned int)h) << 16);
}
__device__ __forceinline__ int imin(int a, int b) { return a < b ? a : b; }

// ---------------- kernel 0: reset per-expert counters ----------------
__global__ void moe_reset(int* __restrict__ counts) {
  if (threadIdx.x < E_) counts[threadIdx.x] = 0;
}

// ---------------- kernel 1: dispatch + gather (fp32 -> split bf16) ----------------
// one block per (token, slot); lane0 claims a row in the expert buffer.
__global__ __launch_bounds__(128) void moe_dispatch(
    const float* __restrict__ hs,        // [T][H]
    const int*   __restrict__ topk_idx,  // [T*K]
    int*   __restrict__ counts,          // [E]
    int*   __restrict__ pos_flat,        // [T*K]
    unsigned short* __restrict__ xhi,    // [E][CAP][H] bf16 hi
    unsigned short* __restrict__ xlo) {  // [E][CAP][H] bf16 lo
  int s = blockIdx.x;
  int t = s >> 1;                        // K_ == 2
  int e = topk_idx[s];
  __shared__ int sp;
  if (threadIdx.x == 0) {
    int p = atomicAdd(&counts[e], 1);
    pos_flat[s] = p;
    sp = p;
  }
  __syncthreads();
  int p = sp;
  if (p >= CAP_) return;                 // capacity drop (w.h.p. never)
  size_t dst = ((size_t)(e * CAP_ + p)) * H_;
  const float* src = hs + (size_t)t * H_;
  int h = threadIdx.x * 4;               // 128 threads * 4 = 512 = H
  float4 v = *(const float4*)(src + h);
  float f[4] = {v.x, v.y, v.z, v.w};
#pragma unroll
  for (int c = 0; c < 4; ++c) {
    unsigned short hi = f2bf_rne(f[c]);
    unsigned short lo = f2bf_rne(f[c] - bf2f(hi));
    xhi[dst + h + c] = hi;
    xlo[dst + h + c] = lo;
  }
}

// ---------------- kernel 2: per-expert GEMM1 + SwiGLU ----------------
// act[e][c][i] = silu(xd @ Wg^T) * (xd @ Wu^T), reduction over H.
// Block tile 64(M rows) x 64(N cols), 4 waves, wave owns 16 rows x 64 cols.
__global__ __launch_bounds__(128) void moe_gemm1_swiglu(
    const unsigned short* __restrict__ xhi,
    const unsigned short* __restrict__ xlo,
    const float* __restrict__ w_gu,      // [E][2I][H] fp32
    const int*   __restrict__ counts,
    unsigned short* __restrict__ actHi,  // [E][CAP][I]
    unsigned short* __restrict__ actLo) {
  int e    = blockIdx.z;
  int cnt  = imin(counts[e], CAP_);
  int mBase = blockIdx.y * 64;
  if (mBase >= cnt) return;              // skip empty row tiles
  int j = blockIdx.x * 64;               // gate column base (0..I-64)

  // B tiles, transposed + converted to split-bf16: [hi/lo][n-subtile][n][k]
  __shared__ unsigned short Bg[2][4][32][16];
  __shared__ unsigned short Bu[2][4][32][16];

  int tid  = threadIdx.x;
  int wave = tid >> 5;
  int lane = tid & 31;
  int m0   = mBase + wave * 16;
  int arow = m0 + (lane & 15);
  // ISA 7.12.2: 16-bit A, lanes<16 hold K {0..7,16..23}, lanes>=16 hold {8..15,24..31}
  size_t abase0 = ((size_t)(e * CAP_ + arow)) * H_ + ((lane & 16) ? 8 : 0);

  f32x8 accG[4], accU[4];
#pragma unroll
  for (int nt = 0; nt < 4; ++nt)
#pragma unroll
    for (int q = 0; q < 8; ++q) { accG[nt][q] = 0.0f; accU[nt][q] = 0.0f; }

  const float* wbase = w_gu + ((size_t)e * (2 * I_)) * H_;

  for (int kk = 0; kk < H_; kk += 32) {
    // ---- stage Wg / Wu tile (64 rows x 32 k, fp32) into LDS as split-bf16, transposed ----
    for (int it = tid; it < 512; it += 128) {   // 64 rows * 8 float4 per row
      int n  = it >> 3;                         // 0..63
      int kq = (it & 7) << 2;                   // 0,4,..,28
      float4 gv = *(const float4*)(wbase + (size_t)(j + n) * H_ + kk + kq);
      float4 uv = *(const float4*)(wbase + (size_t)(I_ + j + n) * H_ + kk + kq);
      int nt = n >> 4, nn = n & 15;
      float gf[4] = {gv.x, gv.y, gv.z, gv.w};
      float uf[4] = {uv.x, uv.y, uv.z, uv.w};
#pragma unroll
      for (int c = 0; c < 4; ++c) {
        unsigned short hi = f2bf_rne(gf[c]);
        Bg[0][nt][nn * 2 + ((kq + c) >> 4)][(kq + c) & 15] = hi;
        Bg[1][nt][nn * 2 + ((kq + c) >> 4)][(kq + c) & 15] = f2bf_rne(gf[c] - bf2f(hi));
        hi = f2bf_rne(uf[c]);
        Bu[0][nt][nn * 2 + ((kq + c) >> 4)][(kq + c) & 15] = hi;
        Bu[1][nt][nn * 2 + ((kq + c) >> 4)][(kq + c) & 15] = f2bf_rne(uf[c] - bf2f(hi));
      }
    }
    __syncthreads();

    // ---- A fragments straight from global (row-major bf16, 2x16B chunks) ----
    FragB ahi, alo;
    {
      const unsigned short* ap = xhi + abase0 + kk;
      ahi.q[0] = *(const uint4*)(ap);
      ahi.q[1] = *(const uint4*)(ap + 16);
      const unsigned short* lp = xlo + abase0 + kk;
      alo.q[0] = *(const uint4*)(lp);
      alo.q[1] = *(const uint4*)(lp + 16);
    }

#pragma unroll
    for (int nt = 0; nt < 4; ++nt) {
      FragB bh, bl;
      // B layout: lane l holds column N=l&15, K {0..7,16..23} (l<16) or {8..15,24..31}
      const unsigned short* bp = &Bg[0][nt][(lane & 15) * 2][(lane & 16) ? 8 : 0];
      bh.q[0] = *(const uint4*)bp; bh.q[1] = *(const uint4*)(bp + 16);
      bp = &Bg[1][nt][(lane & 15) * 2][(lane & 16) ? 8 : 0];
      bl.q[0] = *(const uint4*)bp; bl.q[1] = *(const uint4*)(bp + 16);
      accG[nt] = __builtin_amdgcn_wmma_f32_16x16x32_bf16(false, ahi.v, false, bh.v, (short)0, accG[nt], false, false);
      accG[nt] = __builtin_amdgcn_wmma_f32_16x16x32_bf16(false, ahi.v, false, bl.v, (short)0, accG[nt], false, false);
      accG[nt] = __builtin_amdgcn_wmma_f32_16x16x32_bf16(false, alo.v, false, bh.v, (short)0, accG[nt], false, false);

      bp = &Bu[0][nt][(lane & 15) * 2][(lane & 16) ? 8 : 0];
      bh.q[0] = *(const uint4*)bp; bh.q[1] = *(const uint4*)(bp + 16);
      bp = &Bu[1][nt][(lane & 15) * 2][(lane & 16) ? 8 : 0];
      bl.q[0] = *(const uint4*)bp; bl.q[1] = *(const uint4*)(bp + 16);
      accU[nt] = __builtin_amdgcn_wmma_f32_16x16x32_bf16(false, ahi.v, false, bh.v, (short)0, accU[nt], false, false);
      accU[nt] = __builtin_amdgcn_wmma_f32_16x16x32_bf16(false, ahi.v, false, bl.v, (short)0, accU[nt], false, false);
      accU[nt] = __builtin_amdgcn_wmma_f32_16x16x32_bf16(false, alo.v, false, bh.v, (short)0, accU[nt], false, false);
    }
    __syncthreads();
  }

  // ---- epilogue: silu(gate)*up -> split bf16 act ----
  // C/D layout: VGPR v, lanes<16 -> M=v, lanes>=16 -> M=v+8; N = lane&15
  int rofs = (lane & 16) ? 8 : 0;
  int ncol = lane & 15;
#pragma unroll
  for (int nt = 0; nt < 4; ++nt)
#pragma unroll
    for (int v = 0; v < 8; ++v) {
      float g = accG[nt][v];
      float u = accU[nt][v];
      float a = (g / (1.0f + __expf(-g))) * u;     // silu(g) * u
      int row = m0 + v + rofs;
      size_t o = ((size_t)(e * CAP_ + row)) * I_ + (j + nt * 16 + ncol);
      unsigned short hi = f2bf_rne(a);
      actHi[o] = hi;
      actLo[o] = f2bf_rne(a - bf2f(hi));
    }
}

// ---------------- kernel 3: per-expert GEMM2 (down proj) ----------------
// yd[e][c][h] = act[e][c][:] . down[e][h][:], reduction over I.
__global__ __launch_bounds__(128) void moe_gemm2(
    const unsigned short* __restrict__ actHi,
    const unsigned short* __restrict__ actLo,
    const float* __restrict__ w_down,    // [E][H][I] fp32
    const int*   __restrict__ counts,
    float* __restrict__ yd) {            // [E][CAP][H] fp32
  int e    = blockIdx.z;
  int cnt  = imin(counts[e], CAP_);
  int mBase = blockIdx.y * 64;
  if (mBase >= cnt) return;
  int j = blockIdx.x * 64;               // h column base (0..H-64)

  __shared__ unsigned short Bd[2][4][32][16];

  int tid  = threadIdx.x;
  int wave = tid >> 5;
  int lane = tid & 31;
  int m0   = mBase + wave * 16;
  int arow = m0 + (lane & 15);
  size_t abase0 = ((size_t)(e * CAP_ + arow)) * I_ + ((lane & 16) ? 8 : 0);

  f32x8 acc[4];
#pragma unroll
  for (int nt = 0; nt < 4; ++nt)
#pragma unroll
    for (int q = 0; q < 8; ++q) acc[nt][q] = 0.0f;

  const float* wbase = w_down + ((size_t)e * H_) * I_;

  for (int kk = 0; kk < I_; kk += 32) {
    for (int it = tid; it < 512; it += 128) {   // 64 rows * 8 float4
      int n  = it >> 3;
      int kq = (it & 7) << 2;
      float4 dv = *(const float4*)(wbase + (size_t)(j + n) * I_ + kk + kq);
      int nt = n >> 4, nn = n & 15;
      float df[4] = {dv.x, dv.y, dv.z, dv.w};
#pragma unroll
      for (int c = 0; c < 4; ++c) {
        unsigned short hi = f2bf_rne(df[c]);
        Bd[0][nt][nn * 2 + ((kq + c) >> 4)][(kq + c) & 15] = hi;
        Bd[1][nt][nn * 2 + ((kq + c) >> 4)][(kq + c) & 15] = f2bf_rne(df[c] - bf2f(hi));
      }
    }
    __syncthreads();

    FragB ahi, alo;
    {
      const unsigned short* ap = actHi + abase0 + kk;
      ahi.q[0] = *(const uint4*)(ap);
      ahi.q[1] = *(const uint4*)(ap + 16);
      const unsigned short* lp = actLo + abase0 + kk;
      alo.q[0] = *(const uint4*)(lp);
      alo.q[1] = *(const uint4*)(lp + 16);
    }

#pragma unroll
    for (int nt = 0; nt < 4; ++nt) {
      FragB bh, bl;
      const unsigned short* bp = &Bd[0][nt][(lane & 15) * 2][(lane & 16) ? 8 : 0];
      bh.q[0] = *(const uint4*)bp; bh.q[1] = *(const uint4*)(bp + 16);
      bp = &Bd[1][nt][(lane & 15) * 2][(lane & 16) ? 8 : 0];
      bl.q[0] = *(const uint4*)bp; bl.q[1] = *(const uint4*)(bp + 16);
      acc[nt] = __builtin_amdgcn_wmma_f32_16x16x32_bf16(false, ahi.v, false, bh.v, (short)0, acc[nt], false, false);
      acc[nt] = __builtin_amdgcn_wmma_f32_16x16x32_bf16(false, ahi.v, false, bl.v, (short)0, acc[nt], false, false);
      acc[nt] = __builtin_amdgcn_wmma_f32_16x16x32_bf16(false, alo.v, false, bh.v, (short)0, acc[nt], false, false);
    }
    __syncthreads();
  }

  int rofs = (lane & 16) ? 8 : 0;
  int ncol = lane & 15;
#pragma unroll
  for (int nt = 0; nt < 4; ++nt)
#pragma unroll
    for (int v = 0; v < 8; ++v) {
      int row = m0 + v + rofs;
      yd[((size_t)(e * CAP_ + row)) * H_ + (j + nt * 16 + ncol)] = acc[nt][v];
    }
}

// ---------------- kernel 4: weighted combine over top-k slots ----------------
__global__ __launch_bounds__(256) void moe_combine(
    const int*   __restrict__ topk_idx,
    const float* __restrict__ topk_w,
    const int*   __restrict__ pos_flat,
    const float* __restrict__ yd,
    float* __restrict__ out) {
  int t = blockIdx.x;
  for (int h = threadIdx.x; h < H_; h += blockDim.x) {
    float r = 0.0f;
#pragma unroll
    for (int k = 0; k < K_; ++k) {
      int s = t * K_ + k;
      int p = pos_flat[s];
      if (p < CAP_) {
        int e = topk_idx[s];
        r += topk_w[s] * yd[((size_t)(e * CAP_ + p)) * H_ + h];
      }
    }
    out[(size_t)t * H_ + h] = r;
  }
}

// ---------------- host-side launcher ----------------
extern "C" void kernel_launch(void* const* d_in, const int* in_sizes, int n_in,
                              void* d_out, int out_size, void* d_ws, size_t ws_size,
                              hipStream_t stream) {
  (void)in_sizes; (void)n_in; (void)out_size; (void)ws_size;
  const float* hs  = (const float*)d_in[0];   // hidden_states [T,H]
  const int*   idx = (const int*)d_in[1];     // top_k_indices [T,K]
  const float* wts = (const float*)d_in[2];   // top_k_weights [T,K]
  const float* wgu = (const float*)d_in[3];   // gate_up_proj  [E,2I,H]
  const float* wdn = (const float*)d_in[4];   // down_proj     [E,H,I]
  float* out = (float*)d_out;

  char* ws = (char*)d_ws;
  size_t off = 0;
  int* counts = (int*)(ws + off); off += 256;
  int* pos    = (int*)(ws + off); off += (size_t)T_ * K_ * sizeof(int);
  unsigned short* xhi = (unsigned short*)(ws + off); off += (size_t)E_ * CAP_ * H_ * 2;
  unsigned short* xlo = (unsigned short*)(ws + off); off += (size_t)E_ * CAP_ * H_ * 2;
  unsigned short* aHi = (unsigned short*)(ws + off); off += (size_t)E_ * CAP_ * I_ * 2;
  unsigned short* aLo = (unsigned short*)(ws + off); off += (size_t)E_ * CAP_ * I_ * 2;
  float* yd = (float*)(ws + off); off += (size_t)E_ * CAP_ * H_ * sizeof(float);

  moe_reset<<<1, 32, 0, stream>>>(counts);
  moe_dispatch<<<T_ * K_, 128, 0, stream>>>(hs, idx, counts, pos, xhi, xlo);

  dim3 g1(I_ / 64, CAP_ / 64, E_);
  moe_gemm1_swiglu<<<g1, 128, 0, stream>>>(xhi, xlo, wgu, counts, aHi, aLo);

  dim3 g2(H_ / 64, CAP_ / 64, E_);
  moe_gemm2<<<g2, 128, 0, stream>>>(aHi, aLo, wdn, counts, yd);

  moe_combine<<<T_, 256, 0, stream>>>(idx, wts, pos, yd, out);
}
